// GenuineAttention_16432544874498
// MI455X (gfx1250) — matrix operand order repaired and checked by reading, not compiled, not run on hardware
//
#include <hip/hip_runtime.h>

// ---------------------------------------------------------------------------
// GenuineAttention for MI455X (gfx1250, wave32, WMMA)
// fp32 in/out, bf16 internal matmuls via v_wmma_f32_16x16x32_bf16.
// Flash-style fused softmax (2-pass, recompute) to avoid a 536MB score
// round-trip; attn_weights written once (mandatory output).
// GEMM tiles staged to LDS via GLOBAL_LOAD_ASYNC_TO_LDS_B128, double-buffered
// so the async DMA overlaps the WMMA stream (one barrier per K-step).
// Pass-1 softmax keeps lane-local online stats; cross-lane butterfly once.
// Unroll factors capped to keep the code footprint I-cache friendly.
// ---------------------------------------------------------------------------

#define D_MODEL 1024
#define SEQ     2048
#define N_HEADS 16
#define HDIM    64
#define LOG2E_F 1.44269504f

typedef __attribute__((ext_vector_type(16))) __bf16 v16bf;
typedef __attribute__((ext_vector_type(8)))  float  v8f;
typedef int v4i __attribute__((vector_size(16)));

union FragBF { v16bf v; unsigned int u[8]; uint4 q[2]; };

#if defined(__HIP_DEVICE_COMPILE__) &&                                   \
    __has_builtin(__builtin_amdgcn_global_load_async_to_lds_b128) &&     \
    __has_builtin(__builtin_amdgcn_s_wait_asynccnt)
#define USE_ASYNC_LDS 1
#endif

__device__ __forceinline__ void async_cp16(const void* g, void* l) {
#ifdef USE_ASYNC_LDS
  __builtin_amdgcn_global_load_async_to_lds_b128(
      (__attribute__((address_space(1))) v4i*)(v4i*)(g),
      (__attribute__((address_space(3))) v4i*)(v4i*)(l), 0, 0);
#else
  (void)g; (void)l;
#endif
}

__device__ __forceinline__ void wait_async0() {
#ifdef USE_ASYNC_LDS
  __builtin_amdgcn_s_wait_asynccnt(0);
#endif
}

__device__ __forceinline__ unsigned short f2bf(float f) {
  unsigned int u = __float_as_uint(f);
  u += 0x7FFFu + ((u >> 16) & 1u);          // round-to-nearest-even
  return (unsigned short)(u >> 16);
}

// ---------------------------------------------------------------------------
__global__ void cvt_f32_bf16(const float* __restrict__ src,
                             unsigned short* __restrict__ dst, int n) {
  int i = blockIdx.x * blockDim.x + threadIdx.x;
  if (i < n) dst[i] = f2bf(src[i]);
}

// ---------------------------------------------------------------------------
// C[m,n] = sum_k A[m,k] * W[n,k]   (A: 2048x1024 bf16, W: 1024x1024 bf16)
// MODE 0/1: fused RoPE, store bf16 [h][m][d] (Q / K)
// MODE 2  : store bf16 transposed [h][d][m]   (V, for contiguous B-operand)
// MODE 3  : store fp32 row-major to dst_f32   (final @ wo.T)
// Block: 256 threads (8 waves). Tile: 128(M) x 64(N). Wave: 16 rows x 64 cols.
template <int MODE>
__global__ __launch_bounds__(256)
void gemm_xwT(const unsigned short* __restrict__ A,
              const unsigned short* __restrict__ W,
              unsigned short* __restrict__ dst_bf,
              float* __restrict__ dst_f32,
              const float* __restrict__ cosp,
              const float* __restrict__ sinp) {
  __shared__ __align__(16) unsigned int lds_a[2][128 * 16];  // 128 x 32 bf16
  __shared__ __align__(16) unsigned int lds_b[2][64 * 16];   //  64 x 32 bf16

  const int m0   = blockIdx.x * 128;
  const int n0   = blockIdx.y * 64;
  const int tid  = threadIdx.x;
  const int lane = tid & 31;
  const int wave = tid >> 5;
  const int lg   = lane >> 4;                // lane half-group
  const int ll   = lane & 15;
  const int kb_dw  = lg * 4;                 // A-frag dword base (K 0/8)
  const int kb2_dw = lg * 8;                 // dword base for K 0/16 (unused here)

  v8f acc[4] = {};                           // 4 x (16x16 f32)

  const unsigned int* Adw = (const unsigned int*)A;
  const unsigned int* Wdw = (const unsigned int*)W;

  auto issue_tiles = [&](int kk, int half) {
#ifdef USE_ASYNC_LDS
    // A tile: 512 x 16B chunks (2/thread); B tile: 256 x 16B (1/thread)
#pragma unroll
    for (int i = 0; i < 2; ++i) {
      int c = tid + 256 * i;
      int row = c >> 2, ch = c & 3;
      async_cp16(&A[(size_t)(m0 + row) * D_MODEL + kk + ch * 8],
                 &lds_a[half][row * 16 + ch * 4]);
    }
    {
      int row = tid >> 2, ch = tid & 3;
      async_cp16(&W[(size_t)(n0 + row) * D_MODEL + kk + ch * 8],
                 &lds_b[half][row * 16 + ch * 4]);
    }
#else
#pragma unroll
    for (int i = 0; i < 8; ++i) {            // A tile: 2048 dwords
      int id = tid + 256 * i;
      int row = id >> 4, cp = id & 15;
      lds_a[half][id] = Adw[(size_t)(m0 + row) * (D_MODEL / 2) + (kk >> 1) + cp];
    }
#pragma unroll
    for (int i = 0; i < 4; ++i) {            // B tile: 1024 dwords
      int id = tid + 256 * i;
      int row = id >> 4, cp = id & 15;
      lds_b[half][id] = Wdw[(size_t)(n0 + row) * (D_MODEL / 2) + (kk >> 1) + cp];
    }
#endif
  };

  // software pipeline: prefetch next K-slice while computing current
  issue_tiles(0, 0);
  wait_async0();
  __syncthreads();

  int cur = 0;
#pragma unroll 2
  for (int kk = 0; kk < D_MODEL; kk += 32) {
    if (kk + 32 < D_MODEL) issue_tiles(kk + 32, cur ^ 1);

    FragBF af;                               // A 16x32, ISA 16-bit layout
    int arow = wave * 16 + ll;
#pragma unroll
    for (int j = 0; j < 4; ++j) af.u[j]     = lds_a[cur][arow * 16 + kb_dw + j];
#pragma unroll
    for (int j = 0; j < 4; ++j) af.u[4 + j] = lds_a[cur][arow * 16 + 8 + kb_dw + j];

#pragma unroll
    for (int t = 0; t < 4; ++t) {
      FragBF bf;                             // B 32x16, ISA layout
      int brow = t * 16 + ll;
#pragma unroll
      for (int j = 0; j < 8; ++j) bf.u[j] = lds_b[cur][brow * 16 + kb_dw + (j & 3) + ((j >> 2) << 3)];
      acc[t] = __builtin_amdgcn_wmma_f32_16x16x32_bf16(
          false, af.v, false, bf.v, (short)0, acc[t], false, false);
    }
    wait_async0();
    __syncthreads();
    cur ^= 1;
  }

  // --- store (D layout: VGPR r, lane -> row r+8*lg, col ll) ---
#pragma unroll
  for (int t = 0; t < 4; ++t) {
#pragma unroll
    for (int r = 0; r < 8; ++r) {
      int mrow = m0 + wave * 16 + r + 8 * lg;
      int n    = n0 + t * 16 + ll;
      float val = acc[t][r];
      if constexpr (MODE <= 1) {             // Q/K with fused RoPE
        float partner = __shfl_xor(val, 1, 32);   // pair col sits in lane^1
        int   p = (n & 63) >> 1;
        float c = cosp[mrow * 32 + p];
        float s = sinp[mrow * 32 + p];
        float res = (n & 1) ? (partner * s + val * c)   // o1 = x0*s + x1*c
                            : (val * c - partner * s);  // o0 = x0*c - x1*s
        int h = n >> 6, d = n & 63;
        dst_bf[(size_t)h * SEQ * HDIM + (size_t)mrow * HDIM + d] = f2bf(res);
      } else if constexpr (MODE == 2) {      // V transposed [h][d][m]
        int h = n >> 6, d = n & 63;
        dst_bf[(size_t)h * HDIM * SEQ + (size_t)d * SEQ + mrow] = f2bf(val);
      } else {                               // final projection, fp32
        dst_f32[(size_t)mrow * D_MODEL + n] = val;
      }
    }
  }
}

// ---------------------------------------------------------------------------
// Fused attention: per (head, 128-row tile). 8 waves x 16 rows.
// Pass 1: lane-local online max/sumexp over all 2048 keys (WMMA scores),
//         single cross-lane butterfly at the end.
// Pass 2: recompute scores, emit softmax probs (fp32, mandatory output),
//         entropy accumulation, and P@V via LDS transpose -> WMMA.
__global__ __launch_bounds__(256)
void attn_fused(const unsigned short* __restrict__ qb,
                const unsigned short* __restrict__ kb,
                const unsigned short* __restrict__ vtb,
                unsigned short* __restrict__ aob,
                float* __restrict__ attn_w,   // [h][q][k]
                float* __restrict__ ent_out)  // [q][h]
{
  __shared__ __align__(16) unsigned int lds_p_dw[8 * 16 * 16]; // 8 x 16x32 bf16
  unsigned short* lds_p = (unsigned short*)lds_p_dw;
  const uint4* ldsp4 = (const uint4*)lds_p_dw;

  const int mt = blockIdx.x, h = blockIdx.y;
  const int tid  = threadIdx.x;
  const int lane = tid & 31, wave = tid >> 5;
  const int lg = lane >> 4, ll = lane & 15;
  const int kb_dw = lg * 4, kb2_dw = lg * 8;
  const int m0w = mt * 128 + wave * 16;

  const uint4* q4 = (const uint4*)qb;
  const uint4* k4 = (const uint4*)kb;
  const uint4* v4 = (const uint4*)vtb;
  const int wbase_us = wave * 16 * 32;
  const int wbase_dw = wave * 16 * 16;

  // Q fragments (d = 0..31, 32..63) held in registers for both passes
  FragBF qa[2];
  {
    size_t base = (size_t)h * 65536 + (size_t)(m0w + ll) * 32;
#pragma unroll
    for (int s2 = 0; s2 < 2; ++s2) {
      size_t b = base + s2 * 16 + kb2_dw;     // dword index, %4 == 0
      qa[s2].q[0] = q4[b >> 2];
      qa[s2].q[1] = q4[(b >> 2) + 1];
    }
  }

  // ---- pass 1: lane-local online max / sumexp (no cross-lane per tile) ----
  float lmax[8], lsum[8];
#pragma unroll
  for (int r = 0; r < 8; ++r) { lmax[r] = -1e30f; lsum[r] = 0.f; }

#pragma unroll 2
  for (int nc = 0; nc < SEQ; nc += 16) {
    v8f sacc = {};
#pragma unroll
    for (int s2 = 0; s2 < 2; ++s2) {
      FragBF bfr;
      size_t b = (size_t)h * 65536 + (size_t)(nc + ll) * 32 + s2 * 16 + kb2_dw;
      bfr.q[0] = k4[b >> 2];
      bfr.q[1] = k4[(b >> 2) + 1];
      sacc = __builtin_amdgcn_wmma_f32_16x16x32_bf16(
          false, qa[s2].v, false, bfr.v, (short)0, sacc, false, false);
    }
#pragma unroll
    for (int r = 0; r < 8; ++r) {
      float s = sacc[r] * 0.125f;            // HEAD_DIM^-0.5
      float mnew = fmaxf(lmax[r], s);
      lsum[r] = lsum[r] * __expf(lmax[r] - mnew) + __expf(s - mnew);
      lmax[r] = mnew;
    }
  }

  // single cross-lane butterfly: exact global max / sum per row
  float rmax[8], invZ[8], lz[8], tacc[8];
#pragma unroll
  for (int r = 0; r < 8; ++r) {
    float gm = lmax[r];
#pragma unroll
    for (int msk = 1; msk <= 8; msk <<= 1) gm = fmaxf(gm, __shfl_xor(gm, msk, 32));
    float gs = lsum[r] * __expf(lmax[r] - gm);
#pragma unroll
    for (int msk = 1; msk <= 8; msk <<= 1) gs += __shfl_xor(gs, msk, 32);
    rmax[r] = gm;
    invZ[r] = 1.f / gs;
    lz[r]   = __logf(gs);
    tacc[r] = 0.f;
  }

  v8f oacc[4] = {};

  // ---- pass 2: probs out + entropy + P@V ----
#pragma unroll 1
  for (int nc = 0; nc < SEQ; nc += 32) {
#pragma unroll
    for (int t = 0; t < 2; ++t) {
      int ncol = nc + t * 16;
      v8f sacc = {};
#pragma unroll
      for (int s2 = 0; s2 < 2; ++s2) {
        FragBF bfr;
        size_t b = (size_t)h * 65536 + (size_t)(ncol + ll) * 32 + s2 * 16 + kb2_dw;
        bfr.q[0] = k4[b >> 2];
        bfr.q[1] = k4[(b >> 2) + 1];
        sacc = __builtin_amdgcn_wmma_f32_16x16x32_bf16(
            false, qa[s2].v, false, bfr.v, (short)0, sacc, false, false);
      }
#pragma unroll
      for (int r = 0; r < 8; ++r) {
        float s  = sacc[r] * 0.125f;
        float sv = s - rmax[r];
        float p  = __expf(sv) * invZ[r];
        int mrow = m0w + r + 8 * lg;
        attn_w[(size_t)h * SEQ * SEQ + (size_t)mrow * SEQ + ncol + ll] = p;
        tacc[r] += p * sv;                   // entropy = (logZ - sum p*sv)*log2e
        lds_p[wbase_us + (r + 8 * lg) * 32 + t * 16 + ll] = f2bf(p);
      }
    }
    __syncthreads();
    FragBF pf;                               // P 16x32 as A-fragment
    {
      int b = wbase_dw + ll * 16 + kb2_dw;   // %4 == 0
      pf.q[0] = ldsp4[b >> 2];
      pf.q[1] = ldsp4[(b >> 2) + 1];
    }
#pragma unroll
    for (int t = 0; t < 4; ++t) {
      FragBF vf;                             // V^T rows contiguous along seq
      size_t b = (size_t)h * 65536 + (size_t)(t * 16 + ll) * 1024 + (nc >> 1) + kb2_dw;
      vf.q[0] = v4[b >> 2];
      vf.q[1] = v4[(b >> 2) + 1];
      oacc[t] = __builtin_amdgcn_wmma_f32_16x16x32_bf16(
          false, pf.v, false, vf.v, (short)0, oacc[t], false, false);
    }
    __syncthreads();
  }

  // entropy (one lane per half-group writes)
#pragma unroll
  for (int r = 0; r < 8; ++r) {
    float t = tacc[r];
#pragma unroll
    for (int msk = 1; msk <= 8; msk <<= 1) t += __shfl_xor(t, msk, 32);
    if (ll == 0) {
      int mrow = m0w + r + 8 * lg;
      ent_out[(size_t)mrow * N_HEADS + h] = (lz[r] - t) * LOG2E_F;
    }
  }
  // context -> bf16 [m][h*64+d] for final projection
#pragma unroll
  for (int t = 0; t < 4; ++t)
#pragma unroll
    for (int r = 0; r < 8; ++r) {
      int mrow = m0w + r + 8 * lg;
      int n = h * HDIM + t * 16 + ll;
      aob[(size_t)mrow * D_MODEL + n] = f2bf(oacc[t][r]);
    }
}

// ---------------------------------------------------------------------------
extern "C" void kernel_launch(void* const* d_in, const int* in_sizes, int n_in,
                              void* d_out, int out_size, void* d_ws, size_t ws_size,
                              hipStream_t stream) {
  (void)in_sizes; (void)n_in; (void)out_size; (void)ws_size;
  const float* x    = (const float*)d_in[0];
  const float* wq   = (const float*)d_in[1];
  const float* wk   = (const float*)d_in[2];
  const float* wv   = (const float*)d_in[3];
  const float* wo   = (const float*)d_in[4];
  const float* fcos = (const float*)d_in[5];
  const float* fsin = (const float*)d_in[6];
  float* out = (float*)d_out;

  // workspace layout (bf16 elements); total 14,680,064 elems = 28 MB
  unsigned short* ws = (unsigned short*)d_ws;
  const size_t XB  = 0;          // x bf16            2048x1024
  const size_t WQB = 2097152;    // wq bf16           1024x1024
  const size_t WKB = 3145728;
  const size_t WVB = 4194304;
  const size_t WOB = 5242880;
  const size_t QB  = 6291456;    // q [h][m][d]
  const size_t KB  = 8388608;    // k [h][m][d]
  const size_t VTB = 10485760;   // v [h][d][m]
  const size_t AOB = 12582912;   // context bf16 [m][1024]

  // d_out sections (fp32): out | attn_weights | entropy
  const size_t ATTN_OFF = 2097152;              // 2048*1024
  const size_t ENT_OFF  = ATTN_OFF + (size_t)N_HEADS * SEQ * SEQ;

  cvt_f32_bf16<<<(2097152 + 255) / 256, 256, 0, stream>>>(x,  ws + XB,  2097152);
  cvt_f32_bf16<<<(1048576 + 255) / 256, 256, 0, stream>>>(wq, ws + WQB, 1048576);
  cvt_f32_bf16<<<(1048576 + 255) / 256, 256, 0, stream>>>(wk, ws + WKB, 1048576);
  cvt_f32_bf16<<<(1048576 + 255) / 256, 256, 0, stream>>>(wv, ws + WVB, 1048576);
  cvt_f32_bf16<<<(1048576 + 255) / 256, 256, 0, stream>>>(wo, ws + WOB, 1048576);

  dim3 gg(SEQ / 128, D_MODEL / 64);
  gemm_xwT<0><<<gg, 256, 0, stream>>>(ws + XB, ws + WQB, ws + QB,  nullptr, fcos, fsin);
  gemm_xwT<1><<<gg, 256, 0, stream>>>(ws + XB, ws + WKB, ws + KB,  nullptr, fcos, fsin);
  gemm_xwT<2><<<gg, 256, 0, stream>>>(ws + XB, ws + WVB, ws + VTB, nullptr, fcos, fsin);

  attn_fused<<<dim3(SEQ / 128, N_HEADS), 256, 0, stream>>>(
      ws + QB, ws + KB, ws + VTB, ws + AOB, out + ATTN_OFF, out + ENT_OFF);

  gemm_xwT<3><<<gg, 256, 0, stream>>>(ws + AOB, ws + WOB, nullptr, out, fcos, fsin);
}
